// GraphConvo_64201171141279
// MI455X (gfx1250) — matrix operand order repaired and checked by reading, not compiled, not run on hardware
//
#include <hip/hip_runtime.h>
#include <hip/hip_bf16.h>

typedef __attribute__((ext_vector_type(16))) _Float16 v16h;
typedef __attribute__((ext_vector_type(8)))  _Float16 v8h;
typedef __attribute__((ext_vector_type(4)))  _Float16 v4h;
typedef __attribute__((ext_vector_type(8)))  float    v8f;

#define NNODE 256
#define DIM   64
#define NHEAD 4
#define XSTR  72    // xin row stride (f16), padded
#define WSTR  72    // wT row stride
#define HSTR  264   // hT / att row stride (256 + 8)
#define NEGV  (-9.0e15f)
#define SCMIN (-60000.0f)   // keep scores f16-representable (avoid -inf - -inf = NaN)

__device__ __forceinline__ v16h cat8(v8h lo, v8h hi) {
  return __builtin_shufflevector(lo, hi, 0,1,2,3,4,5,6,7,8,9,10,11,12,13,14,15);
}

// A operand (row-major M x K tile, 16x32 f16). ISA layout: lanes 0-15 hold
// K=[0..8)+[16..24), lanes 16-31 hold K=[8..16)+[24..32) of row M=lane&15.
__device__ __forceinline__ v16h ld_matA(const _Float16* tile, int stride, int lane) {
  int r  = lane & 15;
  int hf = lane >> 4;
  const _Float16* p0 = tile + r * stride + hf * 8;
  v8h lo = *(const v8h*)p0;
  v8h hi = *(const v8h*)(p0 + 16);
  return cat8(lo, hi);
}

// B operand (K x N tile, 32x16 f16) loaded from transposed storage Bt[N][K].
// Lanes 0-15: N=lane, K=[0..8)+[16..24); lanes 16-31: N=lane-16, K=[8..16)+[24..32).
__device__ __forceinline__ v16h ld_matB(const _Float16* tileT, int stride, int lane) {
  int n  = lane & 15;
  int hf = lane >> 4;
  const _Float16* p0 = tileT + n * stride + hf * 8;
  v8h lo = *(const v8h*)p0;
  v8h hi = *(const v8h*)(p0 + 16);
  return cat8(lo, hi);
}

__device__ __forceinline__ v8f wmma_f16(v16h a, v16h b, v8f c) {
  return __builtin_amdgcn_wmma_f32_16x16x32_f16(false, a, false, b, (short)0, c, false, false);
}

__global__ __launch_bounds__(256)
void gat3_kernel(const float* __restrict__ x,
                 const int*   __restrict__ adj,
                 const float* __restrict__ adj_we,
                 const float* __restrict__ W1, const float* __restrict__ a1,
                 const float* __restrict__ W2, const float* __restrict__ a2,
                 const float* __restrict__ W3, const float* __restrict__ a3,
                 float* __restrict__ out) {
  __shared__ _Float16 xin[NNODE * XSTR];      // layer input, f16 row-major
  __shared__ _Float16 hT [DIM * HSTR];        // h transposed: [d][node]
  __shared__ _Float16 wT [DIM * WSTR];        // W^T for current head: [d_out][d_in]
  __shared__ _Float16 attbuf[8 * 16 * HSTR];  // per-wave score/prob rows, f16
  __shared__ float ssrc[NNODE];
  __shared__ float sdst[NNODE];

  const int t    = threadIdx.x;
  const int lane = t & 31;
  const int wave = t >> 5;
  const int r    = lane & 15;
  const int hf   = lane >> 4;
  const int bt   = blockIdx.x;            // 0..511 = (b*T + t)

  // ---- stage layer-0 input into LDS as f16 (packed b64 stores) ----
  {
    const float4* xr = (const float4*)(x + (size_t)bt * NNODE * DIM + (size_t)t * DIM);
    #pragma unroll
    for (int dv = 0; dv < DIM / 4; ++dv) {
      float4 v = xr[dv];
      v4h p; p[0] = (_Float16)v.x; p[1] = (_Float16)v.y;
             p[2] = (_Float16)v.z; p[3] = (_Float16)v.w;
      *(v4h*)&xin[t * XSTR + dv * 4] = p;
    }
  }
  __syncthreads();

  for (int layer = 0; layer < 3; ++layer) {
    const float* Wl = (layer == 0) ? W1 : (layer == 1) ? W2 : W3;
    const float* al = (layer == 0) ? a1 : (layer == 1) ? a2 : a3;

    v8f acc[2][4];
    #pragma unroll
    for (int q = 0; q < 2; ++q)
      #pragma unroll
      for (int nt = 0; nt < 4; ++nt)
        acc[q][nt] = (v8f){0.f,0.f,0.f,0.f,0.f,0.f,0.f,0.f};

    for (int head = 0; head < NHEAD; ++head) {
      // ---- stage W[head]^T into LDS f16: 4x b128 loads in flight, then stores ----
      {
        const float4* wp4 = (const float4*)(Wl + head * 4096 + t * 16);
        float4 w0 = wp4[0], w1 = wp4[1], w2 = wp4[2], w3 = wp4[3];
        int k  = t >> 2;            // W row (d_in)
        int nb = (t & 3) * 16;      // W col (d_out) base
        float v[16] = {w0.x,w0.y,w0.z,w0.w, w1.x,w1.y,w1.z,w1.w,
                       w2.x,w2.y,w2.z,w2.w, w3.x,w3.y,w3.z,w3.w};
        #pragma unroll
        for (int i = 0; i < 16; ++i)
          wT[(nb + i) * WSTR + k] = (_Float16)v[i];
      }
      __syncthreads();

      // ---- h = xin @ W : WMMA, store transposed into hT ----
      #pragma unroll
      for (int q = 0; q < 2; ++q) {
        int m0 = (wave * 2 + q) * 16;
        #pragma unroll
        for (int nt = 0; nt < 4; ++nt) {
          int n0 = nt * 16;
          v8f c = (v8f){0.f,0.f,0.f,0.f,0.f,0.f,0.f,0.f};
          #pragma unroll
          for (int k0 = 0; k0 < DIM; k0 += 32) {
            v16h a = ld_matA(xin + m0 * XSTR + k0, XSTR, lane);
            v16h b = ld_matB(wT  + n0 * WSTR + k0, WSTR, lane);
            c = wmma_f16(a, b, c);
          }
          // lane holds (N = n0+r), rows M = m0 + hf*8 + [0..8) -> contiguous in hT row N
          v8h ph;
          #pragma unroll
          for (int g = 0; g < 8; ++g) ph[g] = (_Float16)c[g];
          *(v8h*)&hT[(n0 + r) * HSTR + m0 + hf * 8] = ph;
        }
      }
      __syncthreads();

      // ---- score projections s_src, s_dst (thread t = node t) ----
      {
        float as = 0.f, ad = 0.f;
        #pragma unroll 8
        for (int d = 0; d < DIM; ++d) {
          float hv = (float)hT[d * HSTR + t];
          as += hv * al[head * 128 + d];
          ad += hv * al[head * 128 + 64 + d];
        }
        ssrc[t] = as;
        sdst[t] = ad;
      }
      __syncthreads();

      // ---- attention: each wave owns 2 blocks of 16 rows ----
      _Float16* attw = attbuf + wave * (16 * HSTR);
      #pragma unroll
      for (int q = 0; q < 2; ++q) {
        int m0  = (wave * 2 + q) * 16;
        int row = m0 + r;
        int c0  = hf * 128;               // lane pair (r, r+16) splits the 256 cols
        float si = ssrc[row];
        const int row_off = row * NNODE;

        // pass 1 (only global pass): masked/weighted score -> f16 buffer, track max
        float mx = -3.0e38f;
        #pragma unroll 2
        for (int jj = 0; jj < 32; ++jj) {
          int c = c0 + jj * 4;
          int4   a4 = *(const int4*)  (adj    + row_off + c);
          float4 w4 = *(const float4*)(adj_we + row_off + c);
          float4 sd = *(const float4*)(&sdst[c]);
          float s0 = si + sd.x, s1 = si + sd.y, s2 = si + sd.z, s3 = si + sd.w;
          float e0 = s0 >= 0.f ? s0 : 0.01f * s0;
          float e1 = s1 >= 0.f ? s1 : 0.01f * s1;
          float e2 = s2 >= 0.f ? s2 : 0.01f * s2;
          float e3 = s3 >= 0.f ? s3 : 0.01f * s3;
          float c0v = fmaxf((a4.x > 0 ? e0 : NEGV) * w4.x, SCMIN);
          float c1v = fmaxf((a4.y > 0 ? e1 : NEGV) * w4.y, SCMIN);
          float c2v = fmaxf((a4.z > 0 ? e2 : NEGV) * w4.z, SCMIN);
          float c3v = fmaxf((a4.w > 0 ? e3 : NEGV) * w4.w, SCMIN);
          mx = fmaxf(mx, fmaxf(fmaxf(c0v, c1v), fmaxf(c2v, c3v)));
          v4h p; p[0] = (_Float16)c0v; p[1] = (_Float16)c1v;
                 p[2] = (_Float16)c2v; p[3] = (_Float16)c3v;
          *(v4h*)&attw[r * HSTR + c] = p;
        }
        mx = fmaxf(mx, __shfl_xor(mx, 16, 32));

        // pass 2 (LDS-only): exp(score - max), sum, overwrite with probs
        float s = 0.f;
        #pragma unroll 2
        for (int jj = 0; jj < 32; ++jj) {
          int c = c0 + jj * 4;
          v4h sc4 = *(v4h*)&attw[r * HSTR + c];
          float p0 = __expf((float)sc4[0] - mx);
          float p1 = __expf((float)sc4[1] - mx);
          float p2 = __expf((float)sc4[2] - mx);
          float p3 = __expf((float)sc4[3] - mx);
          s += (p0 + p1) + (p2 + p3);
          v4h p; p[0] = (_Float16)p0; p[1] = (_Float16)p1;
                 p[2] = (_Float16)p2; p[3] = (_Float16)p3;
          *(v4h*)&attw[r * HSTR + c] = p;
        }
        s += __shfl_xor(s, 16, 32);
        float sinv = 1.0f / s;            // lane r holds 1/sum of row m0+r

        // wave-local LDS write -> read fence (LDS is in-order per wave;
        // stop the compiler from hoisting the matrix loads)
        asm volatile("" ::: "memory");
        __builtin_amdgcn_wave_barrier();

        // per-lane row scales for the C layout (row M = m0 + hf*8 + g)
        float rsc[8];
        #pragma unroll
        for (int g = 0; g < 8; ++g) rsc[g] = __shfl(sinv, hf * 8 + g, 32);

        // ---- hp = att @ h : WMMA over K=256 ----
        #pragma unroll
        for (int nt = 0; nt < 4; ++nt) {
          int n0 = nt * 16;
          v8f c = (v8f){0.f,0.f,0.f,0.f,0.f,0.f,0.f,0.f};
          #pragma unroll
          for (int k0 = 0; k0 < NNODE; k0 += 32) {
            v16h a = ld_matA(attw + k0, HSTR, lane);
            v16h b = ld_matB(hT + n0 * HSTR + k0, HSTR, lane);
            c = wmma_f16(a, b, c);
          }
          #pragma unroll
          for (int g = 0; g < 8; ++g) acc[q][nt][g] += c[g] * rsc[g];
        }
      }
      __syncthreads();
    } // heads

    // ---- mean over heads, relu (layers 0,1), write out + feed next layer ----
    const bool dorelu = (layer < 2);
    #pragma unroll
    for (int q = 0; q < 2; ++q) {
      int m0 = (wave * 2 + q) * 16;
      #pragma unroll
      for (int nt = 0; nt < 4; ++nt) {
        int n0 = nt * 16;
        #pragma unroll
        for (int g = 0; g < 8; ++g) {
          int row = m0 + hf * 8 + g;
          int col = n0 + r;
          float v = acc[q][nt][g] * 0.25f;
          if (dorelu) v = fmaxf(v, 0.f);
          out[((size_t)bt * NNODE + row) * 192 + layer * 64 + col] = v;
          if (layer < 2) xin[row * XSTR + col] = (_Float16)v;
        }
      }
    }
    __syncthreads();
  } // layers
}

extern "C" void kernel_launch(void* const* d_in, const int* in_sizes, int n_in,
                              void* d_out, int out_size, void* d_ws, size_t ws_size,
                              hipStream_t stream) {
  const float* x      = (const float*)d_in[0];
  const int*   adj    = (const int*)  d_in[1];
  const float* adj_we = (const float*)d_in[2];
  const float* W1     = (const float*)d_in[3];
  const float* a1     = (const float*)d_in[4];
  const float* W2     = (const float*)d_in[5];
  const float* a2     = (const float*)d_in[6];
  const float* W3     = (const float*)d_in[7];
  const float* a3     = (const float*)d_in[8];
  float* out = (float*)d_out;

  (void)in_sizes; (void)n_in; (void)d_ws; (void)ws_size; (void)out_size;

  dim3 grid(512);   // B*T graphs
  dim3 block(256);  // 8 wave32
  gat3_kernel<<<grid, block, 0, stream>>>(x, adj, adj_we, W1, a1, W2, a2, W3, a3, out);
}
